// PreProcPMT_67430986547410
// MI455X (gfx1250) — matrix operand order repaired and checked by reading, not compiled, not run on hardware
//
#include <hip/hip_runtime.h>
#include <hip/hip_bf16.h>
#include <stdint.h>

typedef _Float16 v16h __attribute__((ext_vector_type(16)));
typedef _Float16 v8h  __attribute__((ext_vector_type(8)));
typedef float    v8f  __attribute__((ext_vector_type(8)));

#define NBINS 80

// ---------- counter-based RNG (stateless, deterministic) ----------
__device__ __forceinline__ unsigned hmix(unsigned x) {
    x ^= x >> 17; x *= 0xed5ad4bbu;
    x ^= x >> 11; x *= 0xac4c1b51u;
    x ^= x >> 15; x *= 0x31848babu;
    x ^= x >> 14;
    return x;
}
__device__ __forceinline__ unsigned rnd3(unsigned a, unsigned b, unsigned c) {
    return hmix(a * 0x9E3779B9u + b * 0x85EBCA77u + c * 0xC2B2AE3Du + 0x165667B1u);
}
__device__ __forceinline__ float u01(unsigned h) {
    // (0,1) open interval -> safe for log()
    return (float)(h >> 8) * 5.9604645e-08f + 2.9802322e-08f;
}
__device__ __forceinline__ float boxmuller(unsigned h1, unsigned h2) {
    float u1 = u01(h1), u2 = u01(h2);
    return sqrtf(-2.0f * __logf(u1)) * __cosf(6.28318530718f * u2);
}
// truncated normal, redrawn outside +/-2 sigma (matches truncnorm(-2,2))
__device__ __forceinline__ float truncn(unsigned trace, unsigned stream, float mean, float sd) {
    float z = 0.0f;
#pragma unroll 1
    for (int i = 0; i < 8; ++i) {
        float c = boxmuller(rnd3(trace, 0x70000u + stream, 2u * i),
                            rnd3(trace, 0x70000u + stream, 2u * i + 1u));
        if (fabsf(c) <= 2.0f) { z = c; break; }
    }
    return mean + sd * z;
}
// Knuth Poisson sampler; lambda <= 4.75 here so 25 iterations is ~1e-12 tail
__device__ __forceinline__ float poissonf(unsigned trace, unsigned bin, float lam) {
    if (lam <= 0.0f) return 0.0f;
    float L = __expf(-lam);
    float p = 1.0f;
    int k = 0;
#pragma unroll 1
    for (int i = 0; i < 25; ++i) {
        p *= u01(rnd3(trace, bin, 0x50000u + i));
        if (p <= L) break;
        ++k;
    }
    return (float)k;
}

// One wave32 per trace; single-wave workgroup (barriers are S_NOP).
__global__ __launch_bounds__(32) void pmt_kernel(const float* __restrict__ x,
                                                 float* __restrict__ out,
                                                 int ntraces) {
    const int trace = blockIdx.x;
    const int lane  = threadIdx.x;

    // rc[k] = counts[79-k] for k<=79, 0 for k=80..111 (causality pad).
    // counts[t-j] == rc[(79-t) + j]; max index used = 110.
    __shared__ __align__(16) _Float16 rc[112];
    __shared__ __align__(16) _Float16 kernh[96];   // kern, zero-padded j=80..95
    __shared__ __align__(16) float    resf[NBINS]; // conv result

    // gfx1250 prefetch path: pull next trace's input toward L2
    if (trace + 1 < ntraces) __builtin_prefetch(x + (size_t)(trace + 1) * NBINS, 0, 1);

    // ---- per-trace pulse-shape randoms (wave-uniform, computed redundantly) ----
    const float gain = truncn(trace, 0, 10.0f, 2.0f);
    const float tau  = truncn(trace, 1, 7.419741616764922f, 1.5f);
    const float sigm = truncn(trace, 2, 0.8f, 0.06f);
    const float inv_tau = 1.0f / tau;
    const float inv_sig = 1.0f / sigm;

    // ---- stage 1: Poisson counts (reversed) + lognormal kernel into LDS (f16) ----
    rc[80 + lane] = (_Float16)0.0f;               // zero pad 80..111
    for (int b = lane; b < 96; b += 32) {         // uniform 3 iterations per lane
        if (b < NBINS) {
            float lam = x[(size_t)trace * NBINS + b] * 0.25f;
            rc[79 - b] = (_Float16)poissonf(trace, b, lam);
            float tim = 1.25f + 2.5f * (float)b;
            float ln  = __logf(tim * inv_tau) * inv_sig;
            kernh[b]  = (_Float16)(gain * __expf(-0.5f * ln * ln));
        } else {
            kernh[b] = (_Float16)0.0f;            // zero pad so B-tiles need no branch
        }
    }
    __syncthreads();

    // ---- stage 2: causal conv as im2col WMMA: res = Toeplitz(counts) x kern ----
    const int halfSel = lane >> 4;   // 0: lanes 0-15, 1: lanes 16-31
    const int mrow    = lane & 15;   // A-matrix row (M)

    // B (32x16 f16): lane needs kern[32*kt + 16*halfSel + h], h=0..15 ->
    // 32 contiguous aligned bytes -> two ds_load_b128 per tile.
    const v8h* kv = (const v8h*)kernh;
    v16h Bt[3];
#pragma unroll
    for (int kt = 0; kt < 3; ++kt) {
        v8h blo = kv[4 * kt + 2 * halfSel];
        v8h bhi = kv[4 * kt + 2 * halfSel + 1];
        Bt[kt] = __builtin_shufflevector(blo, bhi, 0, 1, 2, 3, 4, 5, 6, 7,
                                         8, 9, 10, 11, 12, 13, 14, 15);
    }

#pragma unroll
    for (int m = 0; m < 5; ++m) {                 // 5 output chunks of 16 bins
        v8f c = {0.f, 0.f, 0.f, 0.f, 0.f, 0.f, 0.f, 0.f};
        const int t = 16 * m + mrow;
        const _Float16* rcp = rc + (79 - t);      // single per-lane base address
        const int nkt = ((16 * m + 15) >> 5) + 1; // causality-pruned K tiles: 1,1,2,2,3
#pragma unroll
        for (int kt = 0; kt < 3; ++kt) {
            if (kt < nkt) {                       // folds at compile time
                v16h a;
#pragma unroll
                for (int e = 0; e < 16; ++e) {    // contiguous A layout: K = 16*half + e (matches B)
                    const int K = e + (halfSel << 4);
                    a[e] = rcp[32 * kt + K];      // immediate DS offset; pad gives zeros
                }
                c = __builtin_amdgcn_wmma_f32_16x16x32_f16(
                        false, a, false, Bt[kt], (short)0, c, false, false);
            }
        }
        // D layout: lanes<16 hold rows 0-7, lanes>=16 rows 8-15; all 16 cols identical.
        if ((lane & 15) == 0) {
#pragma unroll
            for (int r = 0; r < 8; ++r) resf[16 * m + 8 * halfSel + r] = c[r];
        }
    }
    __syncthreads();

    // ---- stage 3: late pulse + baseline noise ----
    float tmax   = -3.4e38f;
    int   tfirst = 1 << 20;
    for (int b = lane; b < NBINS; b += 32) {
        float v = resf[b];
        tmax = fmaxf(tmax, v);
        if (v > 0.0f) tfirst = min(tfirst, b);
    }
#pragma unroll
    for (int off = 16; off > 0; off >>= 1) {
        tmax   = fmaxf(tmax, __shfl_xor(tmax, off, 32));
        tfirst = min(tfirst, __shfl_xor(tfirst, off, 32));
    }
    const bool  has_rise = tfirst < NBINS;
    const bool  do_late  = (u01(rnd3(trace, 0x7Fu, 0)) < 0.75f) && has_rise;
    const float bpos = truncn(trace, 3, 21.0f, 7.0f) + (float)tfirst * 2.5f;
    const float ltau = truncn(trace, 4, 7.419741616764922f, 1.5f);
    const float lsig = truncn(trace, 5, 0.8f, 0.06f);
    const float lsc  = truncn(trace, 6, 28.3f / 290.0f, 10.6f / 290.0f);
    const float amp  = do_late ? (lsc * tmax) : 0.0f;
    const float inv_ltau = 1.0f / ltau;
    const float inv_lsig = 1.0f / lsig;

    for (int b = lane; b < NBINS; b += 32) {
        float r  = resf[b];
        float tg = 1.25f + (float)b * (200.0f / 79.0f);  // np.linspace step = 200/79
        float dt = tg - bpos;
        float pa = 0.0f;
        if (dt > 0.0f) {
            float l = __logf(dt * inv_ltau) * inv_lsig;
            pa = __expf(-0.5f * l * l);
        }
        float nz = boxmuller(rnd3(trace, b, 0x60000u), rnd3(trace, b, 0x60001u));
        out[(size_t)trace * NBINS + b] = r + amp * pa + 1.8f * nz;
    }
}

extern "C" void kernel_launch(void* const* d_in, const int* in_sizes, int n_in,
                              void* d_out, int out_size, void* d_ws, size_t ws_size,
                              hipStream_t stream) {
    (void)n_in; (void)out_size; (void)d_ws; (void)ws_size;
    const float* x = (const float*)d_in[0];
    float* out = (float*)d_out;
    const int ntraces = in_sizes[0] / NBINS;   // 262144
    pmt_kernel<<<ntraces, 32, 0, stream>>>(x, out, ntraces);
}
